// TorchRNN_71330816852792
// MI455X (gfx1250) — matrix operand order repaired and checked by reading, not compiled, not run on hardware
//
#include <hip/hip_runtime.h>
#include <hip/hip_bf16.h>
#include <math.h>

// ---------------------------------------------------------------------------
// TorchRNN for MI455X (gfx1250, wave32, WMMA):
//   phase 1: xp = x @ W_ih^T            one bandwidth-bound bf16-WMMA GEMM
//   phase 2: h = tanh(xp_t + h@W_hh^T)  ONE persistent kernel, 1024 steps,
//            W_hh staged in LDS (fragment-swizzled), grid barrier per step.
// ---------------------------------------------------------------------------

typedef __attribute__((ext_vector_type(16))) __bf16 v16bf;
typedef __attribute__((ext_vector_type(8)))  __bf16 v8bf;
typedef __attribute__((ext_vector_type(8)))  float  v8f;

constexpr int B_ = 64;
constexpr int S_ = 1024;
constexpr int I_ = 256;
constexpr int H_ = 512;

constexpr int NBLK = 8;    // persistent blocks (one 64-col N-chunk each)
constexpr int TPB  = 128;  // 4 waves = 4 M-tiles of 16 batch rows

// Workspace layout (bytes):
constexpr size_t OFF_WHH = 0;                               // H*H bf16 = 512 KB
constexpr size_t OFF_WIH = OFF_WHH + (size_t)H_ * H_ * 2;   // H*I bf16 = 256 KB
constexpr size_t OFF_H0  = OFF_WIH + (size_t)H_ * I_ * 2;   // B*H bf16 =  64 KB
constexpr size_t OFF_H1  = OFF_H0  + (size_t)B_ * H_ * 2;   // B*H bf16 =  64 KB
constexpr size_t OFF_CTR = OFF_H1  + (size_t)B_ * H_ * 2;   // barrier counter
// total < 1 MB of scratch

// ---------------------------------------------------------------------------
// Prep: f32 -> bf16 weights (natural [n][k] layout == per-lane-contiguous
// B-fragment layout), zero h ping-pong, zero the grid-barrier counter.
// ---------------------------------------------------------------------------
__global__ void rnn_prep(const float* __restrict__ Wih, const float* __restrict__ Whh,
                         __bf16* __restrict__ wsWih, __bf16* __restrict__ wsWhh,
                         __bf16* __restrict__ h0, __bf16* __restrict__ h1,
                         unsigned* __restrict__ ctr) {
  int i = blockIdx.x * blockDim.x + threadIdx.x;  // grid covers H*H
  if (i < H_ * H_) wsWhh[i] = (__bf16)Whh[i];
  if (i < H_ * I_) wsWih[i] = (__bf16)Wih[i];
  if (i < B_ * H_) { h0[i] = (__bf16)0.0f; h1[i] = (__bf16)0.0f; }
  if (i == 0) *ctr = 0u;
}

// ---------------------------------------------------------------------------
// Phase 1: xp[(b*S+s), h] = sum_i x[(b*S+s), i] * W_ih[h, i]
// M=65536, N=512, K=256. Block = one 16-row M-tile; 8 waves x 64-col chunks,
// 4 WMMA accumulators per wave, A fragment converted f32->bf16 in registers.
// Streams 192 MB total -> HBM-bandwidth-bound; already at roofline.
// ---------------------------------------------------------------------------
__global__ void __launch_bounds__(256)
rnn_xp_wmma(const float* __restrict__ x,      // (B*S, I) f32
            const __bf16* __restrict__ WihB,  // (H, I) bf16 row-major
            float* __restrict__ xp) {         // (B*S, H) f32
  const int lane  = threadIdx.x & 31;
  const int wave  = threadIdx.x >> 5;
  const int mtile = blockIdx.x;
  const bool hi   = lane >= 16;
  const int  ncol = lane & 15;
  const int  row_m = mtile * 16 + ncol;
  const int  ka0  = hi ? 16 : 0;   // A frag: one 16-half run at ka0
  const int  kb0  = hi ? 16 : 0;   // B frag: one 16-half run at kb0
  const int  nbase = wave * 64;

  v8f acc[4] = {};
  for (int k = 0; k < I_; k += 32) {
    const float* ap = x + (size_t)row_m * I_ + k + ka0;
    v16bf a;
#pragma unroll
    for (int i = 0; i < 8; ++i) a[i] = (__bf16)ap[i];
#pragma unroll
    for (int i = 0; i < 8; ++i) a[8 + i] = (__bf16)ap[8 + i];
#pragma unroll
    for (int j = 0; j < 4; ++j) {
      const __bf16* bp = WihB + (size_t)(nbase + j * 16 + ncol) * I_ + k + kb0;
      v16bf b = *(const v16bf*)bp;
      acc[j] = __builtin_amdgcn_wmma_f32_16x16x32_bf16(
          false, a, false, b, (short)0, acc[j], false, false);
    }
  }
  const int mrow0 = mtile * 16 + (hi ? 8 : 0);
#pragma unroll
  for (int j = 0; j < 4; ++j) {
    float* op = xp + (size_t)mrow0 * H_ + nbase + j * 16 + ncol;
#pragma unroll
    for (int i = 0; i < 8; ++i) op[(size_t)i * H_] = acc[j][i];
  }
}

// ---------------------------------------------------------------------------
// Phase 2 (persistent): all 1024 steps in one kernel.
// Block nb owns cols [nb*64, nb*64+64); wave w owns batch rows [w*16, w*16+16).
// W_hh slice (64 frags x 1 KB) staged once in LDS, fragment-swizzled so each
// ds_load_b128 is perfectly bank-distributed. Grid barrier between steps.
// ---------------------------------------------------------------------------
__global__ void __launch_bounds__(TPB)
rnn_scan_wmma(const __bf16* __restrict__ WhhB,  // (H, H) bf16 row-major
              __bf16* __restrict__ h0,          // ping (B,H) bf16 (zeroed)
              __bf16* __restrict__ h1,          // pong
              float* __restrict__ out,          // (B, S, H) f32: xp in / h out
              float* __restrict__ hn,           // B*H f32 tail (h_n)
              unsigned* __restrict__ ctr) {     // grid-barrier counter
  __shared__ __align__(16) char smem[64 * 1024];  // 64 frags x 1 KB

  const int lane  = threadIdx.x & 31;
  const int wave  = threadIdx.x >> 5;        // 0..3 -> M tile
  const int nbase = blockIdx.x * 64;         // this block's N chunk
  const bool hi   = lane >= 16;
  const int  ncol = lane & 15;
  const int  row_m = wave * 16 + ncol;       // A row for this lane
  const int  ka0  = hi ? 16 : 0;
  const int  kb0  = hi ? 16 : 0;

  // ---- Stage this block's W_hh slice into LDS (once) --------------------
  // Fragment f = ktile*4 + j. Per lane 32 B, split 16 B low / 16 B high so
  // every ds_load_b128 spreads over all 64 banks (2 conflict-free cycles).
  for (int f = wave; f < 64; f += 4) {
    const int ktile = f >> 2, j = f & 3;
    const __bf16* src =
        WhhB + (size_t)(nbase + j * 16 + ncol) * H_ + ktile * 32 + kb0;
    v8bf lo = *(const v8bf*)src;
    v8bf hi8 = *(const v8bf*)(src + 8);
    *(v8bf*)(smem + f * 1024 +       lane * 16) = lo;
    *(v8bf*)(smem + f * 1024 + 512 + lane * 16) = hi8;
  }
  __syncthreads();

  // ---- 1024 sequential steps -------------------------------------------
  for (int t = 0; t < S_; ++t) {
    const __bf16* hprev = (t & 1) ? h1 : h0;
    __bf16*       hnext = (t & 1) ? h0 : h1;

    v8f acc[4] = {};
#pragma unroll 4
    for (int ktile = 0; ktile < 16; ++ktile) {
      const __bf16* ap = hprev + (size_t)row_m * H_ + ktile * 32 + ka0;
      v8bf alo = *(const v8bf*)ap;
      v8bf ahi = *(const v8bf*)(ap + 8);
      v16bf a = __builtin_shufflevector(alo, ahi, 0, 1, 2, 3, 4, 5, 6, 7,
                                        8, 9, 10, 11, 12, 13, 14, 15);
#pragma unroll
      for (int j = 0; j < 4; ++j) {
        const char* fb = smem + (ktile * 4 + j) * 1024;
        v8bf blo = *(const v8bf*)(fb +       lane * 16);
        v8bf bhi = *(const v8bf*)(fb + 512 + lane * 16);
        v16bf b = __builtin_shufflevector(blo, bhi, 0, 1, 2, 3, 4, 5, 6, 7,
                                          8, 9, 10, 11, 12, 13, 14, 15);
        acc[j] = __builtin_amdgcn_wmma_f32_16x16x32_bf16(
            false, a, false, b, (short)0, acc[j], false, false);
      }
    }

    // epilogue: v = tanh(acc + xp); write f32 in place + bf16 for next step
    const int brow0 = wave * 16 + (hi ? 8 : 0);
#pragma unroll
    for (int j = 0; j < 4; ++j) {
      const int col = nbase + j * 16 + ncol;
#pragma unroll
      for (int i = 0; i < 8; ++i) {
        const int b = brow0 + i;
        float* xptr = out + (size_t)b * S_ * H_ + (size_t)t * H_ + col;
        float v = tanhf(acc[j][i] + *xptr);
        *xptr = v;
        hnext[(size_t)b * H_ + col] = (__bf16)v;
        if (t == S_ - 1) hn[(size_t)b * H_ + col] = v;
      }
    }

    // ---- grid barrier across the 8 persistent blocks --------------------
    __syncthreads();
    __builtin_amdgcn_fence(__ATOMIC_RELEASE, "agent");
    if (threadIdx.x == 0) {
      __hip_atomic_fetch_add(ctr, 1u, __ATOMIC_RELAXED,
                             __HIP_MEMORY_SCOPE_AGENT);
      const unsigned tgt = (unsigned)NBLK * (unsigned)(t + 1);
      while (__hip_atomic_load(ctr, __ATOMIC_RELAXED,
                               __HIP_MEMORY_SCOPE_AGENT) < tgt)
        __builtin_amdgcn_s_sleep(1);
    }
    __syncthreads();
    __builtin_amdgcn_fence(__ATOMIC_ACQUIRE, "agent");
  }
}

// ---------------------------------------------------------------------------
extern "C" void kernel_launch(void* const* d_in, const int* in_sizes, int n_in,
                              void* d_out, int out_size, void* d_ws, size_t ws_size,
                              hipStream_t stream) {
  const float* x   = (const float*)d_in[0];   // (B, S, I)
  const float* Wih = (const float*)d_in[1];   // (H, I)
  const float* Whh = (const float*)d_in[2];   // (H, H)
  float* out = (float*)d_out;                 // B*S*H output, then B*H h_n

  char* ws = (char*)d_ws;
  __bf16*   wsWhh = (__bf16*)(ws + OFF_WHH);
  __bf16*   wsWih = (__bf16*)(ws + OFF_WIH);
  __bf16*   h0    = (__bf16*)(ws + OFF_H0);
  __bf16*   h1    = (__bf16*)(ws + OFF_H1);
  unsigned* ctr   = (unsigned*)(ws + OFF_CTR);

  // 1) weight conversion + state init (every call -> deterministic replays)
  rnn_prep<<<(H_ * H_ + 255) / 256, 256, 0, stream>>>(Wih, Whh, wsWih, wsWhh,
                                                      h0, h1, ctr);

  // 2) input projection GEMM (xp written straight into d_out)
  rnn_xp_wmma<<<(B_ * S_) / 16, 256, 0, stream>>>(x, wsWih, out);

  // 3) whole recurrence in one persistent cooperative kernel
  float* hn = out + (size_t)B_ * S_ * H_;
  rnn_scan_wmma<<<NBLK, TPB, 0, stream>>>(wsWhh, h0, h1, out, hn, ctr);
}
